// VRNN_8675833938284
// MI455X (gfx1250) — matrix operand order repaired and checked by reading, not compiled, not run on hardware
//
#include <hip/hip_runtime.h>
#include <cstdint>
#include <cstddef>

// Enable CDNA5 async global->LDS copies for the (already-f16) weight tiles.
#define USE_ASYNC_LDS 1

// ---------------- problem constants ----------------
#define T_STEPS 64
#define BATCH   256
#define XDIM    512
#define CDIM    64
#define HDIM    1024
#define ZDIM    128
#define XOUTD   512
#define EPSF    1.1920929e-07f

typedef _Float16 v16h __attribute__((ext_vector_type(16)));
typedef _Float16 h8   __attribute__((ext_vector_type(8)));
typedef _Float16 h4   __attribute__((ext_vector_type(4)));
typedef float    v8f  __attribute__((ext_vector_type(8)));
typedef float    f4   __attribute__((ext_vector_type(4)));

enum Act { ACT_NONE = 0, ACT_RELU = 1, ACT_ELU = 2, ACT_SIGMOID = 3, ACT_SOFTPLUS01 = 4 };

// ---------------- fp32 -> fp16 weight conversion (once per launch) ----------------
__global__ __launch_bounds__(256) void f32_to_f16_kernel(const float* __restrict__ s,
                                                         _Float16* __restrict__ d, int n) {
  int i = (blockIdx.x * 256 + threadIdx.x) * 4;
  if (i < n) {
    f4 v = *(const f4*)(s + i);
    h4 h;
    h.x = (_Float16)v.x; h.y = (_Float16)v.y; h.z = (_Float16)v.z; h.w = (_Float16)v.w;
    *(h4*)(d + i) = h;
  }
}

// ---------------- WMMA GEMM: C[M,N](f32) = act(A[M,K](f32) @ W[N,K]^T(f16) + bias) ----------------
// Block = 256 threads = 8 waves; tile 128(M) x 128(N), BK = 64, double-buffered LDS.
// Wave (wm,wn) = (wave>>1, wave&1) owns rows [wm*32,+32) x cols [wn*64,+64): 2x4 WMMA tiles.
__global__ __launch_bounds__(256) void gemm_wmma_kernel(
    const float* __restrict__ A, const _Float16* __restrict__ W,
    const float* __restrict__ bias, float* __restrict__ C,
    int N, int K, int ldc, int act) {
  __shared__ __align__(16) _Float16 As[2][128][72];  // [m][k], k-contiguous, pad 8 (row=144B)
  __shared__ __align__(16) _Float16 Bs[2][128][72];  // [n][k], k-contiguous (== W tile layout)

  const int tid  = threadIdx.x;
  const int wave = tid >> 5;
  const int lane = tid & 31;
  const int wm   = wave >> 1;   // 0..3
  const int wn   = wave & 1;    // 0..1
  const int lh   = lane & 15;   // lane within half-wave
  const int lhi  = lane >> 4;   // 0 or 1

  const int m0 = blockIdx.y * 128;
  const int n0 = blockIdx.x * 128;

  v8f acc[2][4];
  const v8f vzero = {};
#pragma unroll
  for (int mi = 0; mi < 2; ++mi)
#pragma unroll
    for (int ni = 0; ni < 4; ++ni) acc[mi][ni] = vzero;

  f4 areg[8];
#if !USE_ASYNC_LDS
  h8 wreg[2];
#endif

  // Issue all A-tile global loads into registers (back-to-back issue, single wait later).
  auto issueA = [&](int k0) {
#pragma unroll
    for (int i = 0; i < 8; ++i) {
      int lin = i * 256 + tid;       // float4 index within 128x64 tile
      int r   = lin >> 4;
      int cc  = (lin & 15) << 2;
      areg[i] = *(const f4*)(A + (size_t)(m0 + r) * K + k0 + cc);
    }
  };
  // Convert staged A registers to f16 and store into LDS buffer.
  auto storeA = [&](int buf) {
#pragma unroll
    for (int i = 0; i < 8; ++i) {
      int lin = i * 256 + tid;
      int r   = lin >> 4;
      int cc  = (lin & 15) << 2;
      h4 hv;
      hv.x = (_Float16)areg[i].x; hv.y = (_Float16)areg[i].y;
      hv.z = (_Float16)areg[i].z; hv.w = (_Float16)areg[i].w;
      *(h4*)(&As[buf][r][cc]) = hv;
    }
  };
  // Weight tile: f16 in memory already -> async copy straight into LDS (no VGPR round-trip).
  auto issueW = [&](int k0, int buf) {
#pragma unroll
    for (int i = 0; i < 2; ++i) {
      int lin = i * 256 + tid;       // h8 (16B) unit within 128x64 tile
      int r   = lin >> 3;
      int cc  = (lin & 7) << 3;
#if USE_ASYNC_LDS
      // ISA aperture rule: LDS byte address == low 32 bits of the generic address.
      uint32_t ldsa   = (uint32_t)(uintptr_t)(&Bs[buf][r][cc]);
      const _Float16* g = W + (size_t)(n0 + r) * K + k0 + cc;
      asm volatile("global_load_async_to_lds_b128 %0, %1, off"
                   :: "v"(ldsa), "v"(g)
                   : "memory");
#else
      wreg[i] = *(const h8*)(W + (size_t)(n0 + r) * K + k0 + cc);
#endif
    }
  };
  auto storeW = [&](int buf) {
#if !USE_ASYNC_LDS
#pragma unroll
    for (int i = 0; i < 2; ++i) {
      int lin = i * 256 + tid;
      int r   = lin >> 3;
      int cc  = (lin & 7) << 3;
      *(h8*)(&Bs[buf][r][cc]) = wreg[i];
    }
#else
    (void)buf;
#endif
  };
  auto waitAsync = [&]() {
#if USE_ASYNC_LDS
    asm volatile("s_wait_asynccnt 0x0" ::: "memory");
#endif
  };

  const int nk = K >> 6;

  // prologue: tile 0 into buffer 0
  issueA(0);
  issueW(0, 0);
  storeA(0);
  storeW(0);
  waitAsync();
  __syncthreads();

  for (int kt = 0; kt < nk; ++kt) {
    const int  buf  = kt & 1;
    const bool more = (kt + 1) < nk;
    if (more) {                        // prefetch next tile while WMMAs run on current one
      issueA((kt + 1) << 6);
      issueW((kt + 1) << 6, buf ^ 1);
    }
#pragma unroll
    for (int ks = 0; ks < 64; ks += 32) {
      v16h afrag[2], bfrag[4];
      // A fragment (ISA 7.12.2, 16-bit A 16x32): lanes0-15 K{0..7,16..23}, lanes16-31 K{8..15,24..31}
#pragma unroll
      for (int mi = 0; mi < 2; ++mi) {
        const _Float16* p = &As[buf][wm * 32 + mi * 16 + lh][ks + lhi * 8];
        h8 lo = *(const h8*)p;
        h8 hi = *(const h8*)(p + 16);
        afrag[mi] = __builtin_shufflevector(lo, hi, 0, 1, 2, 3, 4, 5, 6, 7, 8, 9, 10, 11, 12, 13, 14, 15);
      }
      // B fragment: lane n holds column n, same K order as A (lanes0-15: K{0..7,16..23}, lanes16-31: K{8..15,24..31})
#pragma unroll
      for (int ni = 0; ni < 4; ++ni) {
        const _Float16* p = &Bs[buf][wn * 64 + ni * 16 + lh][ks + lhi * 8];
        h8 lo = *(const h8*)p;
        h8 hi = *(const h8*)(p + 16);
        bfrag[ni] = __builtin_shufflevector(lo, hi, 0, 1, 2, 3, 4, 5, 6, 7, 8, 9, 10, 11, 12, 13, 14, 15);
      }
#pragma unroll
      for (int mi = 0; mi < 2; ++mi)
#pragma unroll
        for (int ni = 0; ni < 4; ++ni)
          acc[mi][ni] = __builtin_amdgcn_wmma_f32_16x16x32_f16(
              false, afrag[mi], false, bfrag[ni], (short)0, acc[mi][ni], false, false);
    }
    if (more) {                        // finish staging next tile, then rendezvous
      storeA(buf ^ 1);
      storeW(buf ^ 1);
      waitAsync();
    }
    __syncthreads();
  }

  // epilogue: bias + activation, scatter per C/D layout (VGPR j -> row j + 8*lhi; lane -> col)
#pragma unroll
  for (int mi = 0; mi < 2; ++mi) {
#pragma unroll
    for (int ni = 0; ni < 4; ++ni) {
      int col  = n0 + wn * 64 + ni * 16 + lh;
      float bv = bias ? bias[col] : 0.0f;
#pragma unroll
      for (int j = 0; j < 8; ++j) {
        int row = m0 + wm * 32 + mi * 16 + lhi * 8 + j;
        float v = acc[mi][ni][j] + bv;
        switch (act) {
          case ACT_RELU:       v = fmaxf(v, 0.0f); break;
          case ACT_ELU:        v = (v > 0.0f) ? v : (__expf(v) - 1.0f); break;
          case ACT_SIGMOID:    v = 1.0f / (1.0f + __expf(-v)); break;
          case ACT_SOFTPLUS01: v = ((v > 20.0f) ? v : log1pf(__expf(v))) + 0.01f; break;
          default: break;
        }
        C[(size_t)row * ldc + col] = v;
      }
    }
  }
}

// ---------------- elementwise helpers ----------------
__global__ __launch_bounds__(256) void concat2_kernel(const float* __restrict__ a, int ka,
                                                      const float* __restrict__ b, int kb,
                                                      float* __restrict__ d, int rows) {
  int kt = ka + kb;
  int n  = rows * kt;
  int i  = blockIdx.x * 256 + threadIdx.x;
  if (i >= n) return;
  int r = i / kt, c = i % kt;
  d[i] = (c < ka) ? a[(size_t)r * ka + c] : b[(size_t)r * kb + (c - ka)];
}

__global__ __launch_bounds__(256) void zsample_kernel(const float* __restrict__ noise,
                                                      const float* __restrict__ em,
                                                      const float* __restrict__ es,
                                                      float* __restrict__ z) {
  int i = blockIdx.x * 256 + threadIdx.x;
  if (i < BATCH * ZDIM) z[i] = noise[i] * es[i] + em[i];
}

__global__ __launch_bounds__(256) void gru_kernel(const float* __restrict__ gi,
                                                  const float* __restrict__ gh,
                                                  float* __restrict__ h) {
  int i = blockIdx.x * 256 + threadIdx.x;
  if (i >= BATCH * HDIM) return;
  int b = i / HDIM, u = i % HDIM;
  const float* gib = gi + (size_t)b * 3 * HDIM;
  const float* ghb = gh + (size_t)b * 3 * HDIM;
  float r = 1.0f / (1.0f + __expf(-(gib[u] + ghb[u])));
  float z = 1.0f / (1.0f + __expf(-(gib[HDIM + u] + ghb[HDIM + u])));
  float n = tanhf(gib[2 * HDIM + u] + r * ghb[2 * HDIM + u]);
  h[i] = (1.0f - z) * n + z * h[i];
}

__global__ __launch_bounds__(256) void kld_kernel(const float* __restrict__ em,
                                                  const float* __restrict__ es,
                                                  const float* __restrict__ pm,
                                                  const float* __restrict__ ps,
                                                  float* __restrict__ out) {
  __shared__ float red[256];
  float s = 0.0f;
  for (int i = threadIdx.x; i < BATCH * ZDIM; i += 256) {
    float e = es[i], p = ps[i], dmn = em[i] - pm[i];
    s += 2.0f * __logf(p) - 2.0f * __logf(e) + (e * e + dmn * dmn) / (p * p) - 1.0f;
  }
  red[threadIdx.x] = s;
  __syncthreads();
  for (int o = 128; o > 0; o >>= 1) {
    if (threadIdx.x < o) red[threadIdx.x] += red[threadIdx.x + o];
    __syncthreads();
  }
  if (threadIdx.x == 0) out[0] += 0.5f * red[0];
}

__global__ __launch_bounds__(256) void nll_kernel(const float* __restrict__ x,
                                                  const float* __restrict__ dm,
                                                  float* __restrict__ out) {
  __shared__ float red[256];
  float s = 0.0f;
  for (int i = threadIdx.x; i < BATCH * XOUTD; i += 256) {
    float xv = x[i], m = dm[i];
    s += xv * __logf(m + EPSF) + (1.0f - xv) * __logf(1.0f - m + EPSF);
  }
  red[threadIdx.x] = s;
  __syncthreads();
  for (int o = 128; o > 0; o >>= 1) {
    if (threadIdx.x < o) red[threadIdx.x] += red[threadIdx.x + o];
    __syncthreads();
  }
  if (threadIdx.x == 0) out[1] += -red[0];
}

__global__ __launch_bounds__(256) void init_kernel(float* __restrict__ h0,
                                                   float* __restrict__ h1,
                                                   float* __restrict__ out01) {
  int i = blockIdx.x * 256 + threadIdx.x;
  if (i < BATCH * HDIM) { h0[i] = 0.0f; h1[i] = 0.0f; }
  if (i < 2) out01[i] = 0.0f;
}

// ---------------- host orchestration ----------------
static inline size_t align256(size_t x) { return (x + 255) & ~(size_t)255; }

extern "C" void kernel_launch(void* const* d_in, const int* in_sizes, int n_in,
                              void* d_out, int out_size, void* d_ws, size_t ws_size,
                              hipStream_t stream) {
  (void)in_sizes; (void)n_in; (void)out_size; (void)ws_size;

  const float* x     = (const float*)d_in[0];
  const float* c     = (const float*)d_in[1];
  const float* noise = (const float*)d_in[2];
  float* out = (float*)d_out;

  // weight specs: {d_in index, N(dout), K(din)}; biases at idx+1 for the first 14
  struct WSpec { int idx, n, k; };
  const WSpec ws[18] = {
      {3, HDIM, XDIM + CDIM},  // phi_x1
      {5, HDIM, HDIM},         // phi_x2
      {7, HDIM, ZDIM},         // phi_z
      {9, HDIM, 2 * HDIM},     // enc1
      {11, HDIM, HDIM},        // enc2
      {13, ZDIM, HDIM},        // enc_mean
      {15, ZDIM, HDIM},        // enc_std
      {17, HDIM, HDIM},        // prior
      {19, ZDIM, HDIM},        // prior_mean
      {21, ZDIM, HDIM},        // prior_std
      {23, HDIM, 2 * HDIM},    // dec1
      {25, HDIM, HDIM},        // dec2
      {27, XOUTD, HDIM},       // dec_mean
      {29, XOUTD, HDIM},       // dec_std
      {31, 3 * HDIM, 2 * HDIM},// gru_ih l0
      {32, 3 * HDIM, HDIM},    // gru_ih l1
      {33, 3 * HDIM, HDIM},    // gru_hh l0
      {34, 3 * HDIM, HDIM},    // gru_hh l1
  };

  char* wsb  = (char*)d_ws;
  size_t off = 0;
  auto alloc_h = [&](size_t e) { _Float16* p = (_Float16*)(wsb + off); off = align256(off + e * 2); return p; };
  auto alloc_f = [&](size_t e) { float* p = (float*)(wsb + off); off = align256(off + e * 4); return p; };

  _Float16* wh[18];
  for (int i = 0; i < 18; ++i) {
    size_t n = (size_t)ws[i].n * ws[i].k;
    wh[i] = alloc_h(n);
    int nb = (int)(n / 4 + 255) / 256;
    f32_to_f16_kernel<<<nb, 256, 0, stream>>>((const float*)d_in[ws[i].idx], wh[i], (int)n);
  }

  float* xin    = alloc_f((size_t)BATCH * (XDIM + CDIM));
  float* t1     = alloc_f((size_t)BATCH * HDIM);
  float* phi_x  = alloc_f((size_t)BATCH * HDIM);
  float* cat_xh = alloc_f((size_t)BATCH * 2 * HDIM);
  float* ench1  = alloc_f((size_t)BATCH * HDIM);
  float* ench   = alloc_f((size_t)BATCH * HDIM);
  float* pri    = alloc_f((size_t)BATCH * HDIM);
  float* pm     = alloc_f((size_t)BATCH * ZDIM);
  float* ps     = alloc_f((size_t)BATCH * ZDIM);
  float* zbuf   = alloc_f((size_t)BATCH * ZDIM);
  float* phiz   = alloc_f((size_t)BATCH * HDIM);
  float* cat_zh = alloc_f((size_t)BATCH * 2 * HDIM);
  float* dech1  = alloc_f((size_t)BATCH * HDIM);
  float* dech   = alloc_f((size_t)BATCH * HDIM);
  float* cat_xz = alloc_f((size_t)BATCH * 2 * HDIM);
  float* gi     = alloc_f((size_t)BATCH * 3 * HDIM);
  float* gh     = alloc_f((size_t)BATCH * 3 * HDIM);
  float* h0     = alloc_f((size_t)BATCH * HDIM);
  float* h1     = alloc_f((size_t)BATCH * HDIM);

  init_kernel<<<(BATCH * HDIM + 255) / 256, 256, 0, stream>>>(h0, h1, out);

  auto gemm = [&](const float* A, const _Float16* W, const float* bias, float* C,
                  int N, int K, int ldc, int act) {
    dim3 grid(N / 128, BATCH / 128);
    gemm_wmma_kernel<<<grid, 256, 0, stream>>>(A, W, bias, C, N, K, ldc, act);
  };
  auto bias_of = [&](int widx) { return (const float*)d_in[ws[widx].idx + 1]; };

  const size_t em_base = 2;
  const size_t es_base = em_base + (size_t)T_STEPS * BATCH * ZDIM;
  const size_t dm_base = es_base + (size_t)T_STEPS * BATCH * ZDIM;
  const size_t ds_base = dm_base + (size_t)T_STEPS * BATCH * XOUTD;

  const int NB_CAT_IN = (BATCH * (XDIM + CDIM) + 255) / 256;
  const int NB_CAT_2H = (BATCH * 2 * HDIM + 255) / 256;
  const int NB_Z      = (BATCH * ZDIM + 255) / 256;
  const int NB_H      = (BATCH * HDIM + 255) / 256;

  for (int t = 0; t < T_STEPS; ++t) {
    const float* x_t = x + (size_t)t * BATCH * XDIM;
    const float* c_t = c + (size_t)t * BATCH * CDIM;
    const float* n_t = noise + (size_t)t * BATCH * ZDIM;
    float* em_t = out + em_base + (size_t)t * BATCH * ZDIM;
    float* es_t = out + es_base + (size_t)t * BATCH * ZDIM;
    float* dm_t = out + dm_base + (size_t)t * BATCH * XOUTD;
    float* ds_t = out + ds_base + (size_t)t * BATCH * XOUTD;

    // feature extractor phi_x
    concat2_kernel<<<NB_CAT_IN, 256, 0, stream>>>(x_t, XDIM, c_t, CDIM, xin, BATCH);
    gemm(xin, wh[0], bias_of(0), t1, HDIM, XDIM + CDIM, HDIM, ACT_ELU);
    gemm(t1, wh[1], bias_of(1), phi_x, HDIM, HDIM, HDIM, ACT_RELU);

    // encoder
    concat2_kernel<<<NB_CAT_2H, 256, 0, stream>>>(phi_x, HDIM, h1, HDIM, cat_xh, BATCH);
    gemm(cat_xh, wh[3], bias_of(3), ench1, HDIM, 2 * HDIM, HDIM, ACT_RELU);
    gemm(ench1, wh[4], bias_of(4), ench, HDIM, HDIM, HDIM, ACT_RELU);
    gemm(ench, wh[5], bias_of(5), em_t, ZDIM, HDIM, ZDIM, ACT_NONE);
    gemm(ench, wh[6], bias_of(6), es_t, ZDIM, HDIM, ZDIM, ACT_SOFTPLUS01);

    // prior
    gemm(h1, wh[7], bias_of(7), pri, HDIM, HDIM, HDIM, ACT_RELU);
    gemm(pri, wh[8], bias_of(8), pm, ZDIM, HDIM, ZDIM, ACT_NONE);
    gemm(pri, wh[9], bias_of(9), ps, ZDIM, HDIM, ZDIM, ACT_SOFTPLUS01);

    // sample + phi_z
    zsample_kernel<<<NB_Z, 256, 0, stream>>>(n_t, em_t, es_t, zbuf);
    gemm(zbuf, wh[2], bias_of(2), phiz, HDIM, ZDIM, HDIM, ACT_RELU);

    // decoder
    concat2_kernel<<<NB_CAT_2H, 256, 0, stream>>>(phiz, HDIM, h1, HDIM, cat_zh, BATCH);
    gemm(cat_zh, wh[10], bias_of(10), dech1, HDIM, 2 * HDIM, HDIM, ACT_RELU);
    gemm(dech1, wh[11], bias_of(11), dech, HDIM, HDIM, HDIM, ACT_RELU);
    gemm(dech, wh[12], bias_of(12), dm_t, XOUTD, HDIM, XOUTD, ACT_SIGMOID);
    gemm(dech, wh[13], bias_of(13), ds_t, XOUTD, HDIM, XOUTD, ACT_SOFTPLUS01);

    // stacked GRU (bias-free)
    concat2_kernel<<<NB_CAT_2H, 256, 0, stream>>>(phi_x, HDIM, phiz, HDIM, cat_xz, BATCH);
    gemm(cat_xz, wh[14], nullptr, gi, 3 * HDIM, 2 * HDIM, 3 * HDIM, ACT_NONE);
    gemm(h0, wh[16], nullptr, gh, 3 * HDIM, HDIM, 3 * HDIM, ACT_NONE);
    gru_kernel<<<NB_H, 256, 0, stream>>>(gi, gh, h0);
    gemm(h0, wh[15], nullptr, gi, 3 * HDIM, HDIM, 3 * HDIM, ACT_NONE);
    gemm(h1, wh[17], nullptr, gh, 3 * HDIM, HDIM, 3 * HDIM, ACT_NONE);
    gru_kernel<<<NB_H, 256, 0, stream>>>(gi, gh, h1);

    // losses
    kld_kernel<<<1, 256, 0, stream>>>(em_t, es_t, pm, ps, out);
    nll_kernel<<<1, 256, 0, stream>>>(x_t, dm_t, out);
  }
}
